// Mamba_8641474200045
// MI455X (gfx1250) — matrix operand order repaired and checked by reading, not compiled, not run on hardware
//
#include <hip/hip_runtime.h>

typedef __attribute__((ext_vector_type(16))) __bf16 v16bf;
typedef __attribute__((ext_vector_type(8)))  __bf16 v8bf;
typedef __attribute__((ext_vector_type(8)))  float  v8f;

// ---------------- problem sizes ----------------
constexpr int BATCH  = 4;
constexpr int TLEN   = 4096;
constexpr int DMODEL = 1024;
constexpr int MROWS  = BATCH * TLEN;   // 16384
constexpr int NPROJ  = 2 * DMODEL;     // 2048

// ---------------- GEMM tiling ----------------
constexpr int BM = 128, BN = 128, BK = 32;
constexpr int LDA_SH = BK + 8;   // bf16 elems; 80B row stride keeps 16B alignment
constexpr int LDB_SH = BK + 8;

// ---------------- scan config ----------------
constexpr int NCHUNK = 64;
constexpr int CLEN   = TLEN / NCHUNK;  // 64 = 2^6

__device__ __forceinline__ float sigmoidf_(float x) { return 1.f / (1.f + __expf(-x)); }

// C = A x B (+bias), A/B converted to bf16, f32 accumulate via v_wmma_f32_16x16x32_bf16.
// Double-buffered LDS, one barrier per K-step. bf16 A operand (GEMM2) is staged with
// CDNA5 async global->LDS (global_load_async_to_lds_b128 + s_wait_asynccnt).
template <typename AT, bool IS_PROJ>
__global__ __launch_bounds__(256) void gemm_wmma(
    const AT* __restrict__ Aglob, const float* __restrict__ Bglob,
    const float* __restrict__ bias,
    float* __restrict__ outF, float* __restrict__ vOut, __bf16* __restrict__ gOut,
    int N, int K)
{
    __shared__ __align__(16) __bf16 Ash[2][BM * LDA_SH];
    __shared__ __align__(16) __bf16 Bsh[2][BN * LDB_SH];

    const int tid  = threadIdx.x;
    const int lane = tid & 31;          // wave32
    const int wave = tid >> 5;          // 0..7
    const int m0 = blockIdx.x * BM;
    const int n0 = blockIdx.y * BN;
    const int wm = (wave >> 1) * 32;    // 4 waves along M
    const int wn = (wave & 1) * 64;     // 2 waves along N, 64 cols each
    const int lm = lane & 15;
    const int hi = lane >> 4;

    // A stager: 2 threads/row, 16 k-elems each
    const int am = tid >> 1;
    const int ak = (tid & 1) * 16;
    // B stager (transpose to [n][k]): 128 n x 2 k-groups of 16
    const int bn = tid & 127;
    const int bq = (tid >> 7) * 16;

    float4 ra[4];   // f32 A staging regs
    float  rb[16];  // B staging regs

    auto issueA = [&](int k0, int buf) {
        if constexpr (sizeof(AT) == 4) {
            const float4* p = (const float4*)(Aglob + (size_t)(m0 + am) * K + k0 + ak);
            ra[0] = p[0]; ra[1] = p[1]; ra[2] = p[2]; ra[3] = p[3];
        } else {
            // CDNA5 async DMA: 16B global -> LDS per lane, ASYNCcnt tracked.
#pragma unroll
            for (int j = 0; j < 2; ++j) {
                const int s   = tid * 2 + j;        // 512 segments of 8 halves
                const int row = s >> 2;
                const int kk  = (s & 3) * 8;
                unsigned lds = (unsigned)(size_t)&Ash[buf][row * LDA_SH + kk];
                const void* g = (const void*)(Aglob + (size_t)(m0 + row) * K + k0 + kk);
                asm volatile("global_load_async_to_lds_b128 %0, %1, off"
                             :: "v"(lds), "v"(g) : "memory");
            }
        }
    };
    auto issueB = [&](int k0) {
        const float* src = Bglob + (size_t)(k0 + bq) * N + (n0 + bn);
#pragma unroll
        for (int i = 0; i < 16; ++i) rb[i] = src[(size_t)i * N];
    };
    auto commit = [&](int buf) {
        if constexpr (sizeof(AT) == 4) {
            const float* t = (const float*)ra;
            v8bf h0, h1;
#pragma unroll
            for (int i = 0; i < 8; ++i) { h0[i] = (__bf16)t[i]; h1[i] = (__bf16)t[8 + i]; }
            *(v8bf*)&Ash[buf][am * LDA_SH + ak]     = h0;
            *(v8bf*)&Ash[buf][am * LDA_SH + ak + 8] = h1;
        }
        v8bf b0, b1;
#pragma unroll
        for (int i = 0; i < 8; ++i) { b0[i] = (__bf16)rb[i]; b1[i] = (__bf16)rb[8 + i]; }
        *(v8bf*)&Bsh[buf][bn * LDB_SH + bq]     = b0;
        *(v8bf*)&Bsh[buf][bn * LDB_SH + bq + 8] = b1;
        if constexpr (sizeof(AT) == 2) {
            asm volatile("s_wait_asynccnt 0x0" ::: "memory");  // async A tile landed
        }
    };

    v8f acc[2][4] = {};

    issueA(0, 0);
    issueB(0);
    commit(0);
    __syncthreads();

    const int nk = K / BK;
    for (int kt = 0; kt < nk; ++kt) {
        const int buf  = kt & 1;
        const bool more = (kt + 1) < nk;
        if (more) {                      // overlap next-tile staging with WMMA
            issueA((kt + 1) * BK, buf ^ 1);
            issueB((kt + 1) * BK);
        }
        if (kt + 2 < nk) {               // global_prefetch_b8 two tiles ahead
            __builtin_prefetch(Aglob + (size_t)(m0 + am) * K + (kt + 2) * BK + ak, 0, 1);
            __builtin_prefetch(Bglob + (size_t)((kt + 2) * BK + bq) * N + (n0 + bn), 0, 1);
        }

        // ---- fragments per ISA 16-bit layouts, 8 wmma per wave per K-step ----
        v16bf af[2];
#pragma unroll
        for (int mi = 0; mi < 2; ++mi) {
            const int row = wm + mi * 16 + lm;
            v8bf lo  = *(const v8bf*)&Ash[buf][row * LDA_SH + hi * 8];
            v8bf hi8 = *(const v8bf*)&Ash[buf][row * LDA_SH + hi * 8 + 16];
#pragma unroll
            for (int i = 0; i < 8; ++i) { af[mi][i] = lo[i]; af[mi][i + 8] = hi8[i]; }
        }
#pragma unroll
        for (int ni = 0; ni < 4; ++ni) {
            const int col = wn + ni * 16 + lm;
            v8bf lo  = *(const v8bf*)&Bsh[buf][col * LDB_SH + hi * 8];
            v8bf hi8 = *(const v8bf*)&Bsh[buf][col * LDB_SH + hi * 8 + 16];
            v16bf bfrag;
#pragma unroll
            for (int i = 0; i < 8; ++i) { bfrag[i] = lo[i]; bfrag[i + 8] = hi8[i]; }
#pragma unroll
            for (int mi = 0; mi < 2; ++mi)
                acc[mi][ni] = __builtin_amdgcn_wmma_f32_16x16x32_bf16(
                    false, af[mi], false, bfrag, (short)0, acc[mi][ni], false, false);
        }

        if (more) commit(buf ^ 1);
        __syncthreads();
    }

    // ---- epilogue (C/D layout: VGPR r -> M = r + 8*hi, N = lane%16) ----
#pragma unroll
    for (int mi = 0; mi < 2; ++mi) {
#pragma unroll
        for (int ni = 0; ni < 4; ++ni) {
#pragma unroll
            for (int r = 0; r < 8; ++r) {
                const int mg = m0 + wm + mi * 16 + hi * 8 + r;
                const int ng = n0 + wn + ni * 16 + lm;
                float val = acc[mi][ni][r] + bias[ng];
                if constexpr (IS_PROJ) {
                    if (ng < DMODEL) vOut[(size_t)mg * DMODEL + ng] = val;
                    else gOut[(size_t)mg * DMODEL + (ng - DMODEL)] = (__bf16)sigmoidf_(val);
                } else {
                    outF[(size_t)mg * DMODEL + ng] = val;
                }
            }
        }
    }
}

// ---- phase 1: per-chunk local scan, emit chunk-final h ----
__global__ __launch_bounds__(256) void scan_carry_k(
    const float* __restrict__ v, const float* __restrict__ Ap, float* __restrict__ carr)
{
    int gid = blockIdx.x * blockDim.x + threadIdx.x;
    int d = gid & (DMODEL - 1);
    int c = (gid >> 10) & (NCHUNK - 1);
    int b = gid >> 16;
    float decay = sigmoidf_(Ap[d]);
    const float* vp = v + (size_t)(b * TLEN + c * CLEN) * DMODEL + d;
    float h = 0.f;
#pragma unroll 4
    for (int t = 0; t < CLEN; ++t) h = __builtin_fmaf(decay, h, vp[(size_t)t * DMODEL]);
    carr[((size_t)b * NCHUNK + c) * DMODEL + d] = h;
}

// ---- phase 2: sequential exclusive scan over chunk carries (decay^CLEN combine) ----
__global__ __launch_bounds__(256) void scan_combine_k(
    const float* __restrict__ Ap, float* __restrict__ carr)
{
    int gid = blockIdx.x * blockDim.x + threadIdx.x; // 0..B*D-1
    int d = gid & (DMODEL - 1);
    int b = gid >> 10;
    float decay = sigmoidf_(Ap[d]);
    float dL = decay;
#pragma unroll
    for (int i = 0; i < 6; ++i) dL *= dL;  // decay^64
    float H = 0.f;
    for (int c = 0; c < NCHUNK; ++c) {
        size_t idx = ((size_t)b * NCHUNK + c) * DMODEL + d;
        float local = carr[idx];
        carr[idx] = H;                      // carry-in for chunk c
        H = __builtin_fmaf(dL, H, local);
    }
}

// ---- phase 3: re-scan with carry-in, fuse C*h and gate, emit bf16 y ----
__global__ __launch_bounds__(256) void scan_apply_k(
    const float* __restrict__ v, const __bf16* __restrict__ g,
    const float* __restrict__ Ap, const float* __restrict__ Cp,
    const float* __restrict__ carr, __bf16* __restrict__ y)
{
    int gid = blockIdx.x * blockDim.x + threadIdx.x;
    int d = gid & (DMODEL - 1);
    int c = (gid >> 10) & (NCHUNK - 1);
    int b = gid >> 16;
    float decay = sigmoidf_(Ap[d]);
    float Cd = Cp[d];
    float h = carr[((size_t)b * NCHUNK + c) * DMODEL + d];
    size_t base = (size_t)(b * TLEN + c * CLEN) * DMODEL + d;
#pragma unroll 4
    for (int t = 0; t < CLEN; ++t) {
        size_t o = base + (size_t)t * DMODEL;
        h = __builtin_fmaf(decay, h, v[o]);
        float yy = Cd * h * (float)g[o];
        y[o] = (__bf16)yy;
    }
}

extern "C" void kernel_launch(void* const* d_in, const int* in_sizes, int n_in,
                              void* d_out, int out_size, void* d_ws, size_t ws_size,
                              hipStream_t stream)
{
    (void)in_sizes; (void)n_in; (void)out_size; (void)ws_size;
    const float* x    = (const float*)d_in[0];
    const float* Win  = (const float*)d_in[1];
    const float* bin  = (const float*)d_in[2];
    const float* Ap   = (const float*)d_in[3];
    const float* Cp   = (const float*)d_in[4];
    const float* Wout = (const float*)d_in[5];
    const float* bout = (const float*)d_in[6];
    float* out = (float*)d_out;

    char* ws = (char*)d_ws;
    float*  v_ws = (float*)ws;                              // 64 MiB f32
    __bf16* g_ws = (__bf16*)(ws + ((size_t)64 << 20));      // 32 MiB bf16
    __bf16* y_ws = (__bf16*)(ws + ((size_t)96 << 20));      // 32 MiB bf16
    float*  carr = (float*)(ws + ((size_t)128 << 20));      //  1 MiB f32

    dim3 blk(256);
    gemm_wmma<float, true><<<dim3(MROWS / BM, NPROJ / BN), blk, 0, stream>>>(
        x, Win, bin, nullptr, v_ws, g_ws, NPROJ, DMODEL);
    scan_carry_k<<<(BATCH * NCHUNK * DMODEL) / 256, blk, 0, stream>>>(v_ws, Ap, carr);
    scan_combine_k<<<(BATCH * DMODEL) / 256, blk, 0, stream>>>(Ap, carr);
    scan_apply_k<<<(BATCH * NCHUNK * DMODEL) / 256, blk, 0, stream>>>(v_ws, g_ws, Ap, Cp, carr, y_ws);
    gemm_wmma<__bf16, false><<<dim3(MROWS / BM, DMODEL / BN), blk, 0, stream>>>(
        y_ws, Wout, bout, out, nullptr, nullptr, DMODEL, DMODEL);
}
